// SlidingWindowAttention_32134945309317
// MI455X (gfx1250) — matrix operand order repaired and checked by reading, not compiled, not run on hardware
//
#include <hip/hip_runtime.h>

#define DIMD 1024
#define NHEAD 16
#define HDIM 64
#define WIN 256
#define BB 2
#define LL 2048

typedef __attribute__((ext_vector_type(16))) _Float16 v16h;
typedef __attribute__((ext_vector_type(8)))  _Float16 v8h;
typedef __attribute__((ext_vector_type(8)))  float    v8f;
typedef __attribute__((ext_vector_type(4)))  int      v4i;
typedef __attribute__((ext_vector_type(8)))  int      v8i;
typedef __attribute__((ext_vector_type(4)))  unsigned v4u;

union V16U { v16h v; v8h h[2]; };

#if __has_builtin(__builtin_amdgcn_tensor_load_to_lds) && __has_builtin(__builtin_amdgcn_s_wait_tensorcnt)
#define USE_TDM 1
#else
#define USE_TDM 0
#endif

// ---------------------------------------------------------------- cast fp32 -> fp16
__global__ void cast_f32_f16(const float* __restrict__ s, _Float16* __restrict__ d, int n) {
    int i = blockIdx.x * blockDim.x + threadIdx.x;
    if (i < n) d[i] = (_Float16)s[i];
}

#if USE_TDM
// TDM 2-D f16 tile load: `rows` rows of `cols` elements (row stride `row_stride` elems)
// compacted contiguously into LDS. `tensor_cols` bounds dim0 so OOB columns read as zero.
// D# per ISA ch.8; groups 2/3 carry benign dims (tensor_dim2/3/4=1, tile_dim3/4=0 unused).
// This toolchain's builtin is the 6-arg clang-23 form:
//   (uint32x4 g0, int32x8 g1, int32x4 g2, int32x4 g3, int32x8 extra, i32 cpol)
__device__ __forceinline__ void tdm_load_tile_f16(const _Float16* gsrc, const _Float16* ldst,
                                                  unsigned cols, unsigned rows,
                                                  unsigned tensor_cols, unsigned row_stride) {
    unsigned long long ga = (unsigned long long)(const void*)gsrc;
    unsigned lds_byte = (unsigned)(unsigned long long)(const void*)ldst; // low 32 = LDS offset
    v4u g0;
    g0[0] = 1u;                                           // count=1, user descriptor
    g0[1] = lds_byte;                                     // lds_addr
    g0[2] = (unsigned)(ga & 0xffffffffull);               // global_addr[95:64]
    g0[3] = ((unsigned)(ga >> 32) & 0x01ffffffu)          // global_addr[120:96]
            | 0x80000000u;                                // type=2 (bits 127:126)
    v8i g1;
    g1[0] = 0x00010000;                                   // mask=0, data_size=1 (2B), no pad
    g1[1] = (int)(tensor_cols << 16);                     // tensor_dim0[15:0]  (bits 63:48)
    g1[2] = (int)((tensor_cols >> 16) | (rows << 16));    // tensor_dim0[31:16], tensor_dim1[15:0]
    g1[3] = (int)(cols << 16);                            // tensor_dim1[31:16]=0, tile_dim0
    g1[4] = (int)rows;                                    // tile_dim1, tile_dim2=0
    g1[5] = (int)row_stride;                              // tensor_dim0_stride[31:0]
    g1[6] = 0;                                            // stride0[47:32]=0, dim1_stride lo=0
    g1[7] = 0;
    v4i g2; g2[0] = 1; g2[1] = 1; g2[2] = 0; g2[3] = 0;   // tensor_dim2=1, tensor_dim3=1, tile_dim3=0
    v4i g3; g3[0] = 0; g3[1] = (int)(1u << 16); g3[2] = 0; g3[3] = 0; // tensor_dim4=1, tile_dim4=0
    v8i gx = {};                                          // extra group (zero)
    __builtin_amdgcn_tensor_load_to_lds(g0, g1, g2, g3, gx, 0);
}
#endif

// ---------------------------------------------------------------- QKV projection + RoPE
// One wave32 computes a 32x64 tile of q, k, or v: two A fragments share every B fragment.
__global__ __launch_bounds__(32)
void qkv_rope_wmma(const _Float16* __restrict__ xh,
                   const _Float16* __restrict__ wh,     // [3][D][D]
                   const float*    __restrict__ cosp,   // [L][64]
                   const float*    __restrict__ sinp,
                   _Float16* __restrict__ qh,           // [B,H,L,hd]
                   _Float16* __restrict__ kh,           // [B,H,L,hd]
                   _Float16* __restrict__ vth) {        // [B,H,hd,L]
    const int ln = threadIdx.x;
    const int lm = ln & 15;
    const int hi = ln >> 4;
    int id = blockIdx.x;
    const int mt    = id % ((BB * LL) / 32); id /= ((BB * LL) / 32);
    const int h     = id % NHEAD;            id /= NHEAD;
    const int which = id;                    // 0=q 1=k 2=v

    const _Float16* W = wh + (size_t)which * DIMD * DIMD;
    const size_t arow0 = (size_t)(mt * 32 + lm) * DIMD;
    const size_t arow1 = arow0 + (size_t)16 * DIMD;

    v8f acc[2][4] = {};
    for (int kt = 0; kt < DIMD / 32; ++kt) {
        V16U a0, a1;
        a0.h[0] = *(const v8h*)(xh + arow0 + kt * 32 + hi * 8);
        a0.h[1] = *(const v8h*)(xh + arow0 + kt * 32 + 16 + hi * 8);
        a1.h[0] = *(const v8h*)(xh + arow1 + kt * 32 + hi * 8);
        a1.h[1] = *(const v8h*)(xh + arow1 + kt * 32 + 16 + hi * 8);
        __builtin_prefetch(xh + arow0 + kt * 32 + 32, 0, 3);
        __builtin_prefetch(xh + arow1 + kt * 32 + 32, 0, 3);
#pragma unroll
        for (int nt = 0; nt < 4; ++nt) {
            const int n = h * HDIM + nt * 16 + lm;
            V16U bu;
            bu.h[0] = *(const v8h*)(W + (size_t)n * DIMD + kt * 32 + hi * 8);
            bu.h[1] = *(const v8h*)(W + (size_t)n * DIMD + kt * 32 + 16 + hi * 8);
            v16h b = bu.v;
            acc[0][nt] = __builtin_amdgcn_wmma_f32_16x16x32_f16(
                false, a0.v, false, b, (short)0, acc[0][nt], false, false);
            acc[1][nt] = __builtin_amdgcn_wmma_f32_16x16x32_f16(
                false, a1.v, false, b, (short)0, acc[1][nt], false, false);
        }
    }

    if (which == 2) {
#pragma unroll
        for (int mi = 0; mi < 2; ++mi)
#pragma unroll
            for (int nt = 0; nt < 4; ++nt) {
                const int c = nt * 16 + lm;
#pragma unroll
                for (int r = 0; r < 8; ++r) {
                    const int g = mt * 32 + mi * 16 + r + hi * 8;
                    const int b = g / LL, l = g % LL;
                    vth[((size_t)(b * NHEAD + h) * HDIM + c) * LL + l] = (_Float16)acc[mi][nt][r];
                }
            }
    } else {
        _Float16* dst = (which == 0) ? qh : kh;
#pragma unroll
        for (int mi = 0; mi < 2; ++mi)
#pragma unroll
            for (int r = 0; r < 8; ++r) {
                const int g = mt * 32 + mi * 16 + r + hi * 8;
                const int b = g / LL, l = g % LL;
                float o[4];
#pragma unroll
                for (int nt = 0; nt < 4; ++nt) {
                    const int c = nt * 16 + lm;
                    const float cv = cosp[l * HDIM + c];
                    const float sv = sinp[l * HDIM + c];
                    const float x0 = acc[mi][nt][r];
                    const float xp = acc[mi][(nt + 2) & 3][r];
                    const float rot = (nt < 2) ? -xp : xp;
                    o[nt] = x0 * cv + rot * sv;
                }
#pragma unroll
                for (int nt = 0; nt < 4; ++nt)
                    dst[((size_t)(b * NHEAD + h) * LL + l) * HDIM + nt * 16 + lm] = (_Float16)o[nt];
            }
    }
}

// ---------------------------------------------------------------- sliding-window attention
// One wave32 per (b, h, 16-query tile); flash-style online softmax over 17 key blocks.
// K (16x64, 1-D) and V (64 rows x 32 keys, 2-D strided) tiles are double-buffered into
// LDS by the Tensor Data Mover; sync via s_wait_tensorcnt (TDM completes in order).
__global__ __launch_bounds__(32)
void swa_wmma(const _Float16* __restrict__ qh,
              const _Float16* __restrict__ kh,
              const _Float16* __restrict__ vth,
              _Float16* __restrict__ ctx) {
    __shared__ __align__(32) _Float16 Plds[16 * 32];
#if USE_TDM
    __shared__ __align__(32) _Float16 Klds[2][16 * HDIM];
    __shared__ __align__(32) _Float16 Vlds[2][HDIM * 32];
#endif
    const int ln = threadIdx.x;
    const int lm = ln & 15;
    const int hi = ln >> 4;
    int id = blockIdx.x;
    const int qt = id % (LL / 16); id /= (LL / 16);
    const int h  = id % NHEAD;     id /= NHEAD;
    const int b  = id;
    const size_t bh = (size_t)b * NHEAD + h;

    if (ln < 16) {                       // zero K=16..31 half of P staging (padding keys)
        v16h z = {};
        *(v16h*)(Plds + ln * 32 + 16) = z;
    }

    const size_t qrow = (bh * LL + qt * 16 + lm) * HDIM;
    V16U qa[2];
#pragma unroll
    for (int c = 0; c < 2; ++c) {
        qa[c].h[0] = *(const v8h*)(qh + qrow + c * 32 + hi * 8);
        qa[c].h[1] = *(const v8h*)(qh + qrow + c * 32 + 16 + hi * 8);
    }

    float m_run[8], l_run[8];
#pragma unroll
    for (int r = 0; r < 8; ++r) { m_run[r] = -1e30f; l_run[r] = 0.f; }
    v8f o[4] = {};

    const int kt0 = (qt >= 16) ? qt - 16 : 0;
    const _Float16* kbase = kh  + bh * LL * HDIM;
    const _Float16* vbase = vth + bh * HDIM * LL;
#if USE_TDM
    tdm_load_tile_f16(kbase + (size_t)kt0 * 16 * HDIM, &Klds[kt0 & 1][0],
                      1024, 1, 1024, 1024);
    tdm_load_tile_f16(vbase + (size_t)kt0 * 16, &Vlds[kt0 & 1][0],
                      32, HDIM, (unsigned)(LL - kt0 * 16), LL);
#endif
    for (int kt = kt0; kt <= qt; ++kt) {
        const _Float16* kp;
#if USE_TDM
        if (kt < qt) {   // issue next tiles, then wait for the current pair (in-order TDM)
            tdm_load_tile_f16(kbase + (size_t)(kt + 1) * 16 * HDIM, &Klds[(kt + 1) & 1][0],
                              1024, 1, 1024, 1024);
            tdm_load_tile_f16(vbase + (size_t)(kt + 1) * 16, &Vlds[(kt + 1) & 1][0],
                              32, HDIM, (unsigned)(LL - (kt + 1) * 16), LL);
            __builtin_amdgcn_s_wait_tensorcnt(2);
        } else {
            __builtin_amdgcn_s_wait_tensorcnt(0);
        }
        kp = &Klds[kt & 1][0];
#else
        kp = kbase + (size_t)kt * 16 * HDIM;
#endif
        const int ko = lm * HDIM + hi * 8;
        V16U kb0, kb1;
        kb0.h[0] = *(const v8h*)(kp + ko);
        kb0.h[1] = *(const v8h*)(kp + ko + 16);
        kb1.h[0] = *(const v8h*)(kp + ko + 32);
        kb1.h[1] = *(const v8h*)(kp + ko + 48);

        v8f s = {};
        s = __builtin_amdgcn_wmma_f32_16x16x32_f16(false, qa[0].v, false, kb0.v, (short)0, s, false, false);
        s = __builtin_amdgcn_wmma_f32_16x16x32_f16(false, qa[1].v, false, kb1.v, (short)0, s, false, false);

        const int j = kt * 16 + lm;
        float p[8], al[8];
#pragma unroll
        for (int r = 0; r < 8; ++r) {
            const int i = qt * 16 + r + hi * 8;
            float sv = ((j <= i) && (j + WIN >= i)) ? s[r] * 0.125f : -1e30f;
            float rm = sv;
            rm = fmaxf(rm, __shfl_xor(rm, 1, 32));
            rm = fmaxf(rm, __shfl_xor(rm, 2, 32));
            rm = fmaxf(rm, __shfl_xor(rm, 4, 32));
            rm = fmaxf(rm, __shfl_xor(rm, 8, 32));
            const float nm = fmaxf(m_run[r], rm);
            al[r] = __expf(m_run[r] - nm);
            const float pv = __expf(sv - nm);
            float rs = pv;
            rs += __shfl_xor(rs, 1, 32);
            rs += __shfl_xor(rs, 2, 32);
            rs += __shfl_xor(rs, 4, 32);
            rs += __shfl_xor(rs, 8, 32);
            l_run[r] = l_run[r] * al[r] + rs;
            m_run[r] = nm;
            p[r] = pv;
        }
#pragma unroll
        for (int nt = 0; nt < 4; ++nt)
#pragma unroll
            for (int r = 0; r < 8; ++r)
                o[nt][r] *= al[r];

        // transpose P: C-layout (lane=key) -> A-layout (lane=query) via LDS
#pragma unroll
        for (int r = 0; r < 8; ++r)
            Plds[(r + hi * 8) * 32 + lm] = (_Float16)p[r];
        __syncthreads();
        V16U pa;
        pa.h[0] = *(const v8h*)(Plds + lm * 32 + hi * 16);
        pa.h[1] = *(const v8h*)(Plds + lm * 32 + hi * 16 + 8);
        __syncthreads();

#if USE_TDM
        const _Float16* vp = &Vlds[kt & 1][0];
#pragma unroll
        for (int nt = 0; nt < 4; ++nt) {
            const int vo = (nt * 16 + lm) * 32 + hi * 16;
            V16U vb;
            vb.h[0] = *(const v8h*)(vp + vo);
            vb.h[1] = *(const v8h*)(vp + vo + 8);
            o[nt] = __builtin_amdgcn_wmma_f32_16x16x32_f16(
                false, pa.v, false, vb.v, (short)0, o[nt], false, false);
        }
#else
#pragma unroll
        for (int nt = 0; nt < 4; ++nt) {
            int kb = kt * 16 + hi * 16;
            if (kb > LL - 16) kb = LL - 16;   // clamp; padded A-keys are zero anyway
            v16h vb = *(const v16h*)(vth + (bh * HDIM + nt * 16 + lm) * LL + kb);
            o[nt] = __builtin_amdgcn_wmma_f32_16x16x32_f16(
                false, pa.v, false, vb, (short)0, o[nt], false, false);
        }
#endif
    }

#pragma unroll
    for (int r = 0; r < 8; ++r) {
        const int g = b * LL + qt * 16 + r + hi * 8;
        const float inv = 1.f / l_run[r];
#pragma unroll
        for (int nt = 0; nt < 4; ++nt)
            ctx[(size_t)g * DIMD + h * HDIM + nt * 16 + lm] = (_Float16)(o[nt][r] * inv);
    }
}

// ---------------------------------------------------------------- output projection (fp32)
// One wave32 computes a 32x64 tile.
__global__ __launch_bounds__(32)
void oproj_wmma(const _Float16* __restrict__ ctx,
                const _Float16* __restrict__ woh,
                float* __restrict__ out) {
    const int ln = threadIdx.x;
    const int lm = ln & 15;
    const int hi = ln >> 4;
    int id = blockIdx.x;
    const int mt  = id % ((BB * LL) / 32); id /= ((BB * LL) / 32);
    const int n64 = id;
    const size_t arow0 = (size_t)(mt * 32 + lm) * DIMD;
    const size_t arow1 = arow0 + (size_t)16 * DIMD;

    v8f acc[2][4] = {};
    for (int kt = 0; kt < DIMD / 32; ++kt) {
        V16U a0, a1;
        a0.h[0] = *(const v8h*)(ctx + arow0 + kt * 32 + hi * 8);
        a0.h[1] = *(const v8h*)(ctx + arow0 + kt * 32 + 16 + hi * 8);
        a1.h[0] = *(const v8h*)(ctx + arow1 + kt * 32 + hi * 8);
        a1.h[1] = *(const v8h*)(ctx + arow1 + kt * 32 + 16 + hi * 8);
        __builtin_prefetch(ctx + arow0 + kt * 32 + 32, 0, 3);
        __builtin_prefetch(ctx + arow1 + kt * 32 + 32, 0, 3);
#pragma unroll
        for (int nt = 0; nt < 4; ++nt) {
            const int n = n64 * 64 + nt * 16 + lm;
            V16U bu;
            bu.h[0] = *(const v8h*)(woh + (size_t)n * DIMD + kt * 32 + hi * 8);
            bu.h[1] = *(const v8h*)(woh + (size_t)n * DIMD + kt * 32 + 16 + hi * 8);
            v16h bfrag = bu.v;
            acc[0][nt] = __builtin_amdgcn_wmma_f32_16x16x32_f16(
                false, a0.v, false, bfrag, (short)0, acc[0][nt], false, false);
            acc[1][nt] = __builtin_amdgcn_wmma_f32_16x16x32_f16(
                false, a1.v, false, bfrag, (short)0, acc[1][nt], false, false);
        }
    }
#pragma unroll
    for (int mi = 0; mi < 2; ++mi)
#pragma unroll
        for (int nt = 0; nt < 4; ++nt) {
            const int n = n64 * 64 + nt * 16 + lm;
#pragma unroll
            for (int r = 0; r < 8; ++r)
                out[(size_t)(mt * 32 + mi * 16 + r + hi * 8) * DIMD + n] = acc[mi][nt][r];
        }
}

// ----------------------------------------------------------------
extern "C" void kernel_launch(void* const* d_in, const int* in_sizes, int n_in,
                              void* d_out, int out_size, void* d_ws, size_t ws_size,
                              hipStream_t stream) {
    (void)in_sizes; (void)n_in; (void)out_size; (void)ws_size;
    const float* x    = (const float*)d_in[0];
    const float* cosp = (const float*)d_in[1];
    const float* sinp = (const float*)d_in[2];
    const float* Wq   = (const float*)d_in[3];
    const float* Wk   = (const float*)d_in[4];
    const float* Wv   = (const float*)d_in[5];
    const float* Wo   = (const float*)d_in[6];
    float* out = (float*)d_out;

    const size_t XN = (size_t)BB * LL * DIMD;
    const size_t WN = (size_t)DIMD * DIMD;
    _Float16* ws  = (_Float16*)d_ws;            // ~48 MiB of f16 scratch
    _Float16* xh  = ws;                         // [B*L, D]
    _Float16* wh  = xh + XN;                    // [3][D][D]
    _Float16* woh = wh + 3 * WN;                // [D][D]
    _Float16* qh  = woh + WN;                   // [B,H,L,hd]
    _Float16* kh  = qh + XN;                    // [B,H,L,hd]
    _Float16* vth = kh + XN;                    // [B,H,hd,L]
    _Float16* ctx = vth + XN;                   // [B*L, D]

    const int T = 256;
    cast_f32_f16<<<(int)((XN + T - 1) / T), T, 0, stream>>>(x,  xh,          (int)XN);
    cast_f32_f16<<<(int)((WN + T - 1) / T), T, 0, stream>>>(Wq, wh,          (int)WN);
    cast_f32_f16<<<(int)((WN + T - 1) / T), T, 0, stream>>>(Wk, wh + WN,     (int)WN);
    cast_f32_f16<<<(int)((WN + T - 1) / T), T, 0, stream>>>(Wv, wh + 2 * WN, (int)WN);
    cast_f32_f16<<<(int)((WN + T - 1) / T), T, 0, stream>>>(Wo, woh,         (int)WN);

    qkv_rope_wmma<<<((BB * LL) / 32) * NHEAD * 3, 32, 0, stream>>>(xh, wh, cosp, sinp, qh, kh, vth);
    swa_wmma<<<BB * NHEAD * (LL / 16), 32, 0, stream>>>(qh, kh, vth, ctx);
    oproj_wmma<<<((BB * LL) / 32) * (DIMD / 64), 32, 0, stream>>>(ctx, woh, out);
}
